// DecompModel2_4114578669575
// MI455X (gfx1250) — matrix operand order repaired and checked by reading, not compiled, not run on hardware
//
#include <hip/hip_runtime.h>

typedef __bf16          v16bf  __attribute__((ext_vector_type(16)));
typedef float           v8f    __attribute__((ext_vector_type(8)));
typedef unsigned short  u16v8  __attribute__((ext_vector_type(8)));
typedef unsigned short  u16v16 __attribute__((ext_vector_type(16)));

#define Bb   32
#define Tt   128
#define Pp   4096          /* 64*64, C=1 */
#define Ll   8
#define THRc 0.1f
#define BIGc 1.0e8f
#define W_ONE  ((unsigned short)0x3F80u)   /* bf16 1.0 : pixel uncovered */

__device__ __forceinline__ unsigned short f32_to_bf16_bits(float f) {
    union { float f; unsigned u; } c; c.f = f;
    unsigned u = c.u;
    unsigned r = (u + 0x7FFFu + ((u >> 16) & 1u)) >> 16;   // round-to-nearest-even
    return (unsigned short)r;
}

// ---------------------------------------------------------------------------
// K0: E[b,t,p] = (tpl>THR) * ((x-tpl)^2 - (x-bg)^2) in bf16;
//     w[b,p] = bf16 1.0 (uncovered); used = 0.
// ---------------------------------------------------------------------------
__global__ void k_precompute(const float* __restrict__ x,
                             const float* __restrict__ tpl,
                             const float* __restrict__ bg,
                             unsigned short* __restrict__ E,
                             unsigned short* __restrict__ wbf,
                             int* __restrict__ used) {
    unsigned idx = blockIdx.x * 256u + threadIdx.x;        // < 32*128*4096
    int b   = idx >> 19;                                   // 128*4096 = 2^19
    int rem = idx & ((1u << 19) - 1u);
    int p   = rem & (Pp - 1);
    int t   = rem >> 12;

    float tv  = tpl[idx];
    float xv  = x[b * Pp + p];
    float bgv = bg[p];
    float e = 0.f;
    if (tv > THRc) {
        float d1 = xv - tv;
        float d2 = xv - bgv;
        e = d1 * d1 - d2 * d2;
    }
    E[idx] = f32_to_bf16_bits(e);

    if (t == 0) wbf[b * Pp + p] = W_ONE;
    if (idx < Bb * Tt) used[idx] = 0;
}

// ---------------------------------------------------------------------------
// K1: per-step scoring. 128-thread block = 4 waves; each wave owns one
//     16-template tile of batch b. The 8 KB bf16 w-vector of the batch is
//     staged into LDS (visible vector copy + async global->LDS, identical
//     bytes), then consumed per chunk as the broadcast-column WMMA B operand
//     via ds_load. E streams from L2 as the A operand. Two accumulator
//     chains hide the WMMA->WMMA accumulation hazard.
// ---------------------------------------------------------------------------
__global__ void __launch_bounds__(128)
k_err_wmma(const unsigned short* __restrict__ E,
           const unsigned short* __restrict__ wbf,
           float* __restrict__ err) {
    __shared__ unsigned short wlds[Pp];                    // 8 KB

    int tid  = threadIdx.x;
    int lane = tid & 31;
    int wave = tid >> 5;
    int b    = blockIdx.x >> 1;
    int tile = (blockIdx.x & 1) * 4 + wave;

    // ---- stage w[b][0..4095] (bf16) into LDS ------------------------------
    {
        const unsigned short* wsrc = wbf + b * Pp;
        // exact LDS byte offset of wlds via addrspace(3) ptrtoint (also
        // escapes wlds so the asm "memory" clobber covers it)
        unsigned ldsbase = (unsigned)(unsigned long long)
            (__attribute__((address_space(3))) unsigned short*)&wlds[0];
        #pragma unroll
        for (int it = 0; it < 4; ++it) {
            int o = it * 1024 + tid * 8;                   // ushort units, 16B/thread
            unsigned ldsoff = ldsbase + (unsigned)(o * 2);
            unsigned long long gaddr = (unsigned long long)(const void*)(wsrc + o);
            asm volatile("global_load_async_to_lds_b128 %0, %1, off"
                         :: "v"(ldsoff), "v"(gaddr) : "memory");
            *(u16v8*)&wlds[o] = *(const u16v8*)(wsrc + o); // visible copy, same bytes
        }
        asm volatile("s_wait_asynccnt 0x0" ::: "memory");
    }
    __syncthreads();

    int m = lane & 15;      // A row M held by this lane
    int g = lane >> 4;      // K-group select (ISA 16-bit A/B layouts)
    const unsigned short* Erow = E + ((size_t)(b * Tt + tile * 16 + m)) * Pp;

    v8f acc0 = {};
    v8f acc1 = {};
    for (int pc = 0; pc < Pp / 32; pc += 2) {
        #pragma unroll
        for (int u = 0; u < 2; ++u) {
            int p0 = (pc + u) * 32;
            // A (16x32 bf16): lane holds K = g*8..g*8+7 and 16+g*8..16+g*8+7
            u16v8 lo = *(const u16v8*)(Erow + p0 + g * 8);
            u16v8 hi = *(const u16v8*)(Erow + p0 + 16 + g * 8);
            u16v16 aw;
            #pragma unroll
            for (int i = 0; i < 8; ++i) { aw[i] = lo[i]; aw[i + 8] = hi[i]; }

            // B (32x16 bf16), all 16 cols = w: lane group g holds K = g*8..g*8+7 and 16+g*8..16+g*8+7
            u16v8 b0 = *(const u16v8*)(&wlds[p0 + g * 8]);
            u16v8 b1 = *(const u16v8*)(&wlds[p0 + 16 + g * 8]);
            u16v16 bw;
            #pragma unroll
            for (int i = 0; i < 8; ++i) { bw[i] = b0[i]; bw[i + 8] = b1[i]; }

            v16bf av = __builtin_bit_cast(v16bf, aw);
            v16bf bv = __builtin_bit_cast(v16bf, bw);
            if (u == 0)
                acc0 = __builtin_amdgcn_wmma_f32_16x16x32_bf16(
                           false, av, false, bv, (short)0, acc0, false, false);
            else
                acc1 = __builtin_amdgcn_wmma_f32_16x16x32_bf16(
                           false, av, false, bv, (short)0, acc1, false, false);
        }
        if (pc + 2 < Pp / 32)
            __builtin_prefetch(Erow + pc * 32 + 256, 0, 1);   // ~512 B ahead
    }

    // C/D layout: lane L, VGPR r -> M = r + 8*(L>>4), N = L&15. Take column N==0.
    if (m == 0) {
        int mb = g * 8;
        #pragma unroll
        for (int r = 0; r < 8; ++r)
            err[b * Tt + tile * 16 + mb + r] = acc0[r] + acc1[r];
    }
}

// ---------------------------------------------------------------------------
// K2: per-step argmin (+used mask, empty candidate = 0), write template id and
//     objects[:,l], clear w where the selected mask covers. Block per batch.
// ---------------------------------------------------------------------------
__global__ void __launch_bounds__(256)
k_select(const float* __restrict__ tpl,
         const float* __restrict__ err,
         int* __restrict__ used,
         unsigned short* __restrict__ wbf,
         float* __restrict__ out,
         int l) {
    __shared__ float sv[128];
    __shared__ int   si[128];
    __shared__ int   ssel;

    int t = threadIdx.x;
    int b = blockIdx.x;

    if (t < 128) {
        float v = err[b * Tt + t];
        if (used[b * Tt + t]) v = BIGc;
        sv[t] = v; si[t] = t;
    }
    __syncthreads();
    for (int s = 64; s > 0; s >>= 1) {
        if (t < s) {
            float v1 = sv[t], v2 = sv[t + s];
            int   i1 = si[t], i2 = si[t + s];
            if (v2 < v1 || (v2 == v1 && i2 < i1)) { sv[t] = v2; si[t] = i2; }
        }
        __syncthreads();
    }
    if (t == 0) {
        int sel = (sv[0] < 0.f) ? (si[0] + 1) : 0;   // strict <0: ties go to empty
        ssel = sel;
        if (sel > 0) used[b * Tt + sel - 1] = 1;
        out[(size_t)Bb * Pp + (size_t)Bb * Ll * Pp + b * Ll + l] = (float)sel;
    }
    __syncthreads();

    int sel = ssel;
    const float*    tp  = (sel > 0) ? (tpl + ((size_t)(b * Tt + sel - 1)) * Pp) : nullptr;
    float*          obj = out + (size_t)Bb * Pp + ((size_t)(b * Ll + l)) * Pp;
    unsigned short* wb  = wbf + b * Pp;
    for (int i = 0; i < Pp / 256; ++i) {
        int p = i * 256 + t;
        float ov = (sel > 0) ? tp[p] : 0.f;
        obj[p] = ov;                                  // objects[:,l] = full sel_t
        if (sel > 0 && ov > THRc) wb[p] = 0;          // covered -> weight 0
    }
}

// ---------------------------------------------------------------------------
// K3: reconstruction = compose(objects, bg), objects[:,0] on top.
// ---------------------------------------------------------------------------
__global__ void k_compose(const float* __restrict__ bg, float* __restrict__ out) {
    int idx = blockIdx.x * 256 + threadIdx.x;   // < 32*4096
    int b = idx >> 12;
    int p = idx & (Pp - 1);
    float rec = bg[p];
    #pragma unroll
    for (int l = Ll - 1; l >= 0; --l) {
        float o = out[(size_t)Bb * Pp + ((size_t)(b * Ll + l)) * Pp + p];
        if (o > THRc) rec = o;
    }
    out[idx] = rec;
}

// ---------------------------------------------------------------------------
extern "C" void kernel_launch(void* const* d_in, const int* in_sizes, int n_in,
                              void* d_out, int out_size, void* d_ws, size_t ws_size,
                              hipStream_t stream) {
    const float* x   = (const float*)d_in[0];
    const float* tpl = (const float*)d_in[1];
    const float* bg  = (const float*)d_in[2];
    float* out = (float*)d_out;

    // Workspace layout (≈33.8 MB total):
    //   E    : bf16[32][128][4096]  = 33,554,432 B
    //   wbf  : bf16[32][4096]       =    262,144 B   (1.0 = uncovered, 0 = covered)
    //   err  : f32 [32][128]        =     16,384 B
    //   used : i32 [32][128]        =     16,384 B
    char* ws = (char*)d_ws;
    unsigned short* E    = (unsigned short*)ws;
    unsigned short* wbf  = (unsigned short*)(ws + (size_t)Bb * Tt * Pp * 2);
    float*          err  = (float*)(ws + (size_t)Bb * Tt * Pp * 2 + (size_t)Bb * Pp * 2);
    int*            used = (int*)  (ws + (size_t)Bb * Tt * Pp * 2 + (size_t)Bb * Pp * 2
                                       + (size_t)Bb * Tt * 4);

    k_precompute<<<(Bb * Tt * Pp) / 256, 256, 0, stream>>>(x, tpl, bg, E, wbf, used);

    for (int l = 0; l < Ll; ++l) {
        k_err_wmma<<<Bb * 2, 128, 0, stream>>>(E, wbf, err);
        k_select<<<Bb, 256, 0, stream>>>(tpl, err, used, wbf, out, l);
    }

    k_compose<<<(Bb * Pp) / 256, 256, 0, stream>>>(bg, out);

    (void)in_sizes; (void)n_in; (void)out_size; (void)ws_size;
}
